// GenreAdaptiveLayer_15290083574013
// MI455X (gfx1250) — matrix operand order repaired and checked by reading, not compiled, not run on hardware
//
#include <hip/hip_runtime.h>
#include <hip/hip_bf16.h>

#define B_SZ 4096
#define D_SZ 1024
#define H_SZ 2048
#define G_SZ 10
#define EPS 1e-5f
#define SLOPE 0.2f

typedef __attribute__((ext_vector_type(16))) __bf16 v16bf;
typedef __attribute__((ext_vector_type(8)))  float  v8f;
typedef __attribute__((ext_vector_type(8)))  unsigned short v8s;
typedef __attribute__((ext_vector_type(4)))  unsigned v4u;
typedef __attribute__((ext_vector_type(8)))  int v8i;
typedef __attribute__((ext_vector_type(4)))  int v4i;

union Frag16 { v16bf bf; v8s s[2]; };

__device__ __forceinline__ unsigned short f2bf(float f) {
    unsigned u = __float_as_uint(f);
    unsigned r = (u + 0x7FFFu + ((u >> 16) & 1u)) >> 16;   // round-to-nearest-even
    return (unsigned short)r;
}
__device__ __forceinline__ float bf2f(unsigned short s) {
    return __uint_as_float((unsigned)s << 16);
}
__device__ __forceinline__ float leaky(float v) { return v > 0.f ? v : SLOPE * v; }

// ---------------------------------------------------------------------------
// f32 weights -> bf16 in the WMMA B-fragment lane layout.
//   e = [g][kt][nt][lane(32)][p(16)],  k = kt*32 + (lane>>4)*16 + p,
//   col = nt*16 + (lane&15)   => one wave's B fragment = 32 contiguous B/lane
// ---------------------------------------------------------------------------
__global__ __launch_bounds__(256) void swizzle_w_kernel(
    const float* __restrict__ src, unsigned short* __restrict__ dst,
    int Gn, int K, int N)
{
    size_t total = (size_t)Gn * K * N;
    size_t e = (size_t)blockIdx.x * 256 + threadIdx.x;
    if (e >= total) return;
    int p    = (int)(e & 15);
    int lane = (int)((e >> 4) & 31);
    size_t tile = e >> 9;
    int NT = N >> 4, KT = K >> 5;
    int nt = (int)(tile % NT);
    size_t r = tile / NT;
    int kt = (int)(r % KT);
    int g  = (int)(r / KT);
    int k   = kt * 32 + (lane >> 4) * 16 + p;
    int col = nt * 16 + (lane & 15);
    dst[e] = f2bf(src[((size_t)g * K + k) * N + col]);
}

// ---------------------------------------------------------------------------
// ae[g,h] = emb[g,:] @ Wa1[D:,h] + ba1[h]   (tiny: 10x2048)
// ---------------------------------------------------------------------------
__global__ __launch_bounds__(256) void ae_kernel(
    const float* __restrict__ emb, const float* __restrict__ Wa1,
    const float* __restrict__ ba1, float* __restrict__ ae)
{
    int idx = blockIdx.x * 256 + threadIdx.x;
    if (idx >= G_SZ * H_SZ) return;
    int g = idx / H_SZ, h = idx % H_SZ;
    float s = ba1[h];
    for (int k = 0; k < H_SZ; ++k)
        s += emb[(size_t)g * H_SZ + k] * Wa1[(size_t)(D_SZ + k) * H_SZ + h];
    ae[idx] = s;
}

// ---------------------------------------------------------------------------
// GEMM1: 32-row blocks, 2x2 register tiling (4 wmma per 2A+2B fragments),
// fused bias + LayerNorm + leaky. h staged in LDS as bf16 (128 KB).
// LDS: xt 32x1024 bf16 (64 KB) + hbuf 32x2048 bf16 (128 KB) = 192 KB.
// ---------------------------------------------------------------------------
__global__ __launch_bounds__(256) void gemm1_ln_kernel(
    const float* __restrict__ x, const unsigned short* __restrict__ W1b,
    const float* __restrict__ b1, const float* __restrict__ ln_g,
    const float* __restrict__ ln_b, unsigned short* __restrict__ abuf)
{
    extern __shared__ char smem[];
    unsigned short* xt   = (unsigned short*)smem;                 // 32 x 1024 bf16
    unsigned short* hbuf = (unsigned short*)(smem + 32 * D_SZ * 2); // 32 x 2048 bf16
    __shared__ float s_sum[32], s_sq[32];

    const int b0 = blockIdx.x * 32;
    const int g  = blockIdx.y;
    const int tid = threadIdx.x;

    // stage x tile (32 rows) -> LDS as bf16
    for (int i = 0; i < 32; ++i) {
        int c = tid + 256 * i;             // float4 index, 8192 total
        int row = c >> 8;
        int col4 = c & 255;
        float4 v = ((const float4*)(x + (size_t)(b0 + row) * D_SZ))[col4];
        int e = row * D_SZ + col4 * 4;
        xt[e + 0] = f2bf(v.x); xt[e + 1] = f2bf(v.y);
        xt[e + 2] = f2bf(v.z); xt[e + 3] = f2bf(v.w);
    }
    __syncthreads();

    const int lane = tid & 31, wid = tid >> 5;
    const int row16 = lane & 15, khalf = lane >> 4;
    const int KT = D_SZ / 32;   // 32
    const int NT = H_SZ / 16;   // 128
    const size_t kstep = (size_t)NT * 32 * 16;

    for (int t = 0; t < 8; ++t) {
        int nt0 = wid * 16 + 2 * t;
        v8f a00 = {}, a01 = {}, a10 = {}, a11 = {};
        const unsigned short* bp =
            W1b + ((((size_t)g * KT) * NT + nt0) * 32 + lane) * 16;
        for (int kt = 0; kt < KT; ++kt) {
            Frag16 af0, af1, bf0, bf1;
            int ab0 = row16 * D_SZ + kt * 32 + khalf * 16;
            int ab1 = ab0 + 16 * D_SZ;                 // M-tile 1: rows 16..31
            af0.s[0] = *(const v8s*)(xt + ab0);
            af0.s[1] = *(const v8s*)(xt + ab0 + 8);
            af1.s[0] = *(const v8s*)(xt + ab1);
            af1.s[1] = *(const v8s*)(xt + ab1 + 8);
            bf0.s[0] = *(const v8s*)(bp);
            bf0.s[1] = *(const v8s*)(bp + 8);
            bf1.s[0] = *(const v8s*)(bp + 512);        // next N-tile: +32*16 elems
            bf1.s[1] = *(const v8s*)(bp + 520);
            __builtin_prefetch(bp + kstep, 0, 0);
            bp += kstep;
            a00 = __builtin_amdgcn_wmma_f32_16x16x32_bf16(
                false, af0.bf, false, bf0.bf, (short)0, a00, false, false);
            a01 = __builtin_amdgcn_wmma_f32_16x16x32_bf16(
                false, af0.bf, false, bf1.bf, (short)0, a01, false, false);
            a10 = __builtin_amdgcn_wmma_f32_16x16x32_bf16(
                false, af1.bf, false, bf0.bf, (short)0, a10, false, false);
            a11 = __builtin_amdgcn_wmma_f32_16x16x32_bf16(
                false, af1.bf, false, bf1.bf, (short)0, a11, false, false);
        }
        int col0 = nt0 * 16 + row16;
        float b1v0 = b1[(size_t)g * H_SZ + col0];
        float b1v1 = b1[(size_t)g * H_SZ + col0 + 16];
        #pragma unroll
        for (int i = 0; i < 8; ++i) {
            int r = khalf ? (i + 8) : i;               // C/D layout: lanes>=16 -> M=8..15
            hbuf[r * H_SZ + col0]             = f2bf(a00[i] + b1v0);
            hbuf[r * H_SZ + col0 + 16]        = f2bf(a01[i] + b1v1);
            hbuf[(r + 16) * H_SZ + col0]      = f2bf(a10[i] + b1v0);
            hbuf[(r + 16) * H_SZ + col0 + 16] = f2bf(a11[i] + b1v1);
        }
    }
    __syncthreads();

    if (tid < 32) { s_sum[tid] = 0.f; s_sq[tid] = 0.f; }
    __syncthreads();
    {
        int r = tid >> 3, cc = tid & 7;            // 8 threads per row
        float ls = 0.f, lq = 0.f;
        for (int j = 0; j < H_SZ / 8; ++j) {
            float v = bf2f(hbuf[r * H_SZ + cc + 8 * j]);
            ls += v; lq += v * v;
        }
        atomicAdd(&s_sum[r], ls);
        atomicAdd(&s_sq[r], lq);
    }
    __syncthreads();
    {
        int r = tid >> 3, cc = tid & 7;
        float mu  = s_sum[r] * (1.0f / H_SZ);
        float var = s_sq[r] * (1.0f / H_SZ) - mu * mu;
        float rstd = rsqrtf(var + EPS);
        size_t orow = ((size_t)(b0 + r) * G_SZ + g) * H_SZ;
        for (int j = 0; j < H_SZ / 8; ++j) {
            int h = cc + 8 * j;
            float v = bf2f(hbuf[r * H_SZ + h]);
            float hn = (v - mu) * rstd * ln_g[(size_t)g * H_SZ + h]
                       + ln_b[(size_t)g * H_SZ + h];
            abuf[orow + h] = f2bf(leaky(hn));
        }
    }
}

// ---------------------------------------------------------------------------
// GEMM2: outputs[b,g,d] = a[b,g,:] @ W2[g,:,d] + b2[g,d]
// A-tile (32 x 2048 bf16, row stride G*H, 128 KB) staged by ONE TDM op.
// 8 waves x (2M x 2N tiles) -> 32 rows x 256 cols per block, K = 2048.
// ---------------------------------------------------------------------------
__global__ __launch_bounds__(256) void gemm2_kernel(
    const unsigned short* __restrict__ abuf, const unsigned short* __restrict__ W2b,
    const float* __restrict__ b2, float* __restrict__ outbuf)
{
    extern __shared__ char smem[];
    unsigned short* at = (unsigned short*)smem;           // 32 x 2048 bf16 (128 KB)
    const int b0 = blockIdx.x * 32;
    const int nchunk = blockIdx.y;                        // 0..3
    const int g = blockIdx.z;
    const int tid = threadIdx.x;
    const int lane = tid & 31, wid = tid >> 5;

    // ---- TDM: one DMA op stages the whole 2-D tile (32 rows x 4 KB) ----
    if (wid == 0) {
        unsigned lds_base = __builtin_amdgcn_groupstaticsize();
        unsigned long long ga =
            (unsigned long long)(size_t)(abuf + ((size_t)b0 * G_SZ + g) * H_SZ);
        v4u g0; v8i g1; v4i g2, g3;
        g0[0] = 1u;                                   // count=1, user descriptor
        g0[1] = lds_base;                             // lds_addr
        g0[2] = (unsigned)(ga & 0xFFFFFFFFu);         // global_addr[31:0]
        g0[3] = (unsigned)((ga >> 32) & 0x1FFFFFFu) | (2u << 30);  // addr[56:32] | type=2
        g1[0] = (int)(1u << 16);                      // data_size=1 (2 bytes)
        g1[1] = (int)((unsigned)H_SZ << 16);          // tensor_dim0[15:0] = 2048
        g1[2] = (int)(32u << 16);                     // tensor_dim1[15:0] = 32
        g1[3] = (int)((unsigned)H_SZ << 16);          // tile_dim0 = 2048
        g1[4] = 32;                                   // tile_dim1 = 32
        g1[5] = (int)(G_SZ * H_SZ);                   // tensor_dim0_stride = 20480
        g1[6] = 0; g1[7] = 0;
        g2 = (v4i){0, 0, 0, 0};
        g3 = (v4i){0, 0, 0, 0};
#if __clang_major__ >= 23
        v8i gpad = (v8i){0, 0, 0, 0, 0, 0, 0, 0};
        __builtin_amdgcn_tensor_load_to_lds(g0, g1, g2, g3, gpad, 0);
#else
        __builtin_amdgcn_tensor_load_to_lds(g0, g1, g2, g3, 0);
#endif
        __builtin_amdgcn_s_wait_tensorcnt(0);
    }
    __syncthreads();

    const int row16 = lane & 15, khalf = lane >> 4;
    const int KT = H_SZ / 32;   // 64
    const int NT = D_SZ / 16;   // 64
    const size_t kstep = (size_t)NT * 32 * 16;
    const int nt0 = nchunk * 16 + wid * 2;

    v8f a00 = {}, a01 = {}, a10 = {}, a11 = {};
    const unsigned short* bp =
        W2b + ((((size_t)g * KT) * NT + nt0) * 32 + lane) * 16;
    for (int kt = 0; kt < KT; ++kt) {
        Frag16 af0, af1, bf0, bf1;
        int ab0 = row16 * H_SZ + kt * 32 + khalf * 16;
        int ab1 = ab0 + 16 * H_SZ;
        af0.s[0] = *(const v8s*)(at + ab0);
        af0.s[1] = *(const v8s*)(at + ab0 + 8);
        af1.s[0] = *(const v8s*)(at + ab1);
        af1.s[1] = *(const v8s*)(at + ab1 + 8);
        bf0.s[0] = *(const v8s*)(bp);
        bf0.s[1] = *(const v8s*)(bp + 8);
        bf1.s[0] = *(const v8s*)(bp + 512);
        bf1.s[1] = *(const v8s*)(bp + 520);
        __builtin_prefetch(bp + kstep, 0, 0);
        bp += kstep;
        a00 = __builtin_amdgcn_wmma_f32_16x16x32_bf16(
            false, af0.bf, false, bf0.bf, (short)0, a00, false, false);
        a01 = __builtin_amdgcn_wmma_f32_16x16x32_bf16(
            false, af0.bf, false, bf1.bf, (short)0, a01, false, false);
        a10 = __builtin_amdgcn_wmma_f32_16x16x32_bf16(
            false, af1.bf, false, bf0.bf, (short)0, a10, false, false);
        a11 = __builtin_amdgcn_wmma_f32_16x16x32_bf16(
            false, af1.bf, false, bf1.bf, (short)0, a11, false, false);
    }
    int col0 = nt0 * 16 + row16;
    float b2v0 = b2[(size_t)g * D_SZ + col0];
    float b2v1 = b2[(size_t)g * D_SZ + col0 + 16];
    #pragma unroll
    for (int i = 0; i < 8; ++i) {
        int r = khalf ? (i + 8) : i;
        size_t o0 = ((size_t)(b0 + r) * G_SZ + g) * D_SZ + col0;
        size_t o1 = ((size_t)(b0 + r + 16) * G_SZ + g) * D_SZ + col0;
        outbuf[o0]      = a00[i] + b2v0;
        outbuf[o0 + 16] = a01[i] + b2v1;
        outbuf[o1]      = a10[i] + b2v0;
        outbuf[o1 + 16] = a11[i] + b2v1;
    }
}

// ---------------------------------------------------------------------------
// Attention scores: ah = x@Wax via WMMA (32 rows, 2x2 tiling), ae staged to
// LDS by ASYNC global->LDS DMA overlapped with the matrix phase, then
// scores[b,g] = sum_h leaky(ah + ae[g,h]) * Wa2[h] + ba2.
// Dynamic LDS layout (no static LDS so asm offsets == pointer offsets):
//   [0)     ssc   32x10 f32 (2 KB slot)
//   [2K)    xt    32x1024 bf16 (64 KB)
//   [66K)   hbuf  32x2048 bf16 (128 KB)
//   [194K)  aeLds 10x2048 f32  (80 KB)      total ~274 KB (<= 320 KB/WGP)
// ---------------------------------------------------------------------------
#define AT_SSC   0
#define AT_XT    2048
#define AT_HBUF  (AT_XT + 32 * D_SZ * 2)
#define AT_AE    (AT_HBUF + 32 * H_SZ * 2)
#define AT_SMEM  (AT_AE + G_SZ * H_SZ * 4)

__global__ __launch_bounds__(256) void attn_scores_kernel(
    const float* __restrict__ x, const unsigned short* __restrict__ Waxb,
    const float* __restrict__ ae, const float* __restrict__ Wa2,
    const float* __restrict__ ba2, float* __restrict__ scores)
{
    extern __shared__ char smem[];
    float* ssc = (float*)(smem + AT_SSC);                 // [32][10]
    unsigned short* xt   = (unsigned short*)(smem + AT_XT);
    unsigned short* hbuf = (unsigned short*)(smem + AT_HBUF);
    float* aeLds = (float*)(smem + AT_AE);

    const int b0 = blockIdx.x * 32;
    const int tid = threadIdx.x;

    // stage x tile -> LDS bf16
    for (int i = 0; i < 32; ++i) {
        int c = tid + 256 * i;
        int row = c >> 8;
        int col4 = c & 255;
        float4 v = ((const float4*)(x + (size_t)(b0 + row) * D_SZ))[col4];
        int e = row * D_SZ + col4 * 4;
        xt[e + 0] = f2bf(v.x); xt[e + 1] = f2bf(v.y);
        xt[e + 2] = f2bf(v.z); xt[e + 3] = f2bf(v.w);
    }
    // async DMA of ae (80 KB) into LDS; overlaps the WMMA phase
    {
        unsigned base = __builtin_amdgcn_groupstaticsize() + AT_AE;
        for (int i = 0; i < (G_SZ * H_SZ * 4 / 16) / 256; ++i) {   // 20 chunks/thread
            int c = tid + 256 * i;                 // 16-byte chunk index
            unsigned laddr = base + (unsigned)c * 16u;
            const float* gp = ae + (size_t)c * 4;
            asm volatile("global_load_async_to_lds_b128 %0, %1, off"
                         :: "v"(laddr), "v"(gp) : "memory");
        }
    }
    if (tid < 32 * 10) ssc[tid] = 0.f;
    __syncthreads();

    const int lane = tid & 31, wid = tid >> 5;
    const int row16 = lane & 15, khalf = lane >> 4;
    const int KT = D_SZ / 32;   // 32
    const size_t kstep = (size_t)(H_SZ / 16) * 32 * 16;

    for (int t = 0; t < 8; ++t) {
        int nt0 = wid * 16 + 2 * t;
        v8f a00 = {}, a01 = {}, a10 = {}, a11 = {};
        const unsigned short* bp = Waxb + ((size_t)nt0 * 32 + lane) * 16;
        for (int kt = 0; kt < KT; ++kt) {
            Frag16 af0, af1, bf0, bf1;
            int ab0 = row16 * D_SZ + kt * 32 + khalf * 16;
            int ab1 = ab0 + 16 * D_SZ;
            af0.s[0] = *(const v8s*)(xt + ab0);
            af0.s[1] = *(const v8s*)(xt + ab0 + 8);
            af1.s[0] = *(const v8s*)(xt + ab1);
            af1.s[1] = *(const v8s*)(xt + ab1 + 8);
            bf0.s[0] = *(const v8s*)(bp);
            bf0.s[1] = *(const v8s*)(bp + 8);
            bf1.s[0] = *(const v8s*)(bp + 512);
            bf1.s[1] = *(const v8s*)(bp + 520);
            bp += kstep;
            a00 = __builtin_amdgcn_wmma_f32_16x16x32_bf16(
                false, af0.bf, false, bf0.bf, (short)0, a00, false, false);
            a01 = __builtin_amdgcn_wmma_f32_16x16x32_bf16(
                false, af0.bf, false, bf1.bf, (short)0, a01, false, false);
            a10 = __builtin_amdgcn_wmma_f32_16x16x32_bf16(
                false, af1.bf, false, bf0.bf, (short)0, a10, false, false);
            a11 = __builtin_amdgcn_wmma_f32_16x16x32_bf16(
                false, af1.bf, false, bf1.bf, (short)0, a11, false, false);
        }
        int col0 = nt0 * 16 + row16;
        #pragma unroll
        for (int i = 0; i < 8; ++i) {
            int r = khalf ? (i + 8) : i;
            hbuf[r * H_SZ + col0]             = f2bf(a00[i]);
            hbuf[r * H_SZ + col0 + 16]        = f2bf(a01[i]);
            hbuf[(r + 16) * H_SZ + col0]      = f2bf(a10[i]);
            hbuf[(r + 16) * H_SZ + col0 + 16] = f2bf(a11[i]);
        }
    }
    // wait for the async ae DMA, then make everything visible block-wide
    asm volatile("s_wait_asynccnt 0x0" ::: "memory");
    __syncthreads();

    {
        int r = tid >> 3, cc = tid & 7;            // 8 threads per row
        float s[G_SZ];
        #pragma unroll
        for (int g = 0; g < G_SZ; ++g) s[g] = 0.f;
        for (int j = 0; j < H_SZ / 8; ++j) {
            int h = cc + 8 * j;
            float v = bf2f(hbuf[r * H_SZ + h]);
            float wv = Wa2[h];
            #pragma unroll
            for (int g = 0; g < G_SZ; ++g)
                s[g] += leaky(v + aeLds[g * H_SZ + h]) * wv;
        }
        #pragma unroll
        for (int g = 0; g < G_SZ; ++g) atomicAdd(&ssc[r * 10 + g], s[g]);
    }
    __syncthreads();
    for (int idx = tid; idx < 32 * 10; idx += 256) {
        int r = idx / 10, g = idx % 10;
        scores[(size_t)(b0 + r) * G_SZ + g] = ssc[idx] + ba2[0];
    }
}

// ---------------------------------------------------------------------------
// Softmax over G (fused) + weighted sum of outputs -> final (B, D)
// ---------------------------------------------------------------------------
__global__ __launch_bounds__(256) void combine_kernel(
    const float* __restrict__ scores, const float* __restrict__ outbuf,
    float* __restrict__ out)
{
    const int b = blockIdx.x, tid = threadIdx.x;
    float sc[G_SZ];
    float m = -1e30f;
    #pragma unroll
    for (int g = 0; g < G_SZ; ++g) { sc[g] = scores[(size_t)b * G_SZ + g]; m = fmaxf(m, sc[g]); }
    float sum = 0.f;
    #pragma unroll
    for (int g = 0; g < G_SZ; ++g) { sc[g] = __expf(sc[g] - m); sum += sc[g]; }
    float inv = 1.f / sum;
    for (int it = 0; it < D_SZ / 256; ++it) {
        int d = tid + 256 * it;
        float acc = 0.f;
        #pragma unroll
        for (int g = 0; g < G_SZ; ++g)
            acc += outbuf[((size_t)b * G_SZ + g) * D_SZ + d] * sc[g];
        out[(size_t)b * D_SZ + d] = acc * inv;
    }
}

// ---------------------------------------------------------------------------
extern "C" void kernel_launch(void* const* d_in, const int* in_sizes, int n_in,
                              void* d_out, int out_size, void* d_ws, size_t ws_size,
                              hipStream_t stream) {
    const float* x    = (const float*)d_in[0];
    const float* emb  = (const float*)d_in[1];
    const float* W1   = (const float*)d_in[2];
    const float* b1   = (const float*)d_in[3];
    const float* ln_g = (const float*)d_in[4];
    const float* ln_b = (const float*)d_in[5];
    const float* W2   = (const float*)d_in[6];
    const float* b2   = (const float*)d_in[7];
    const float* Wa1  = (const float*)d_in[8];
    const float* ba1  = (const float*)d_in[9];
    const float* Wa2  = (const float*)d_in[10];
    const float* ba2  = (const float*)d_in[11];
    float* out = (float*)d_out;

    char* ws = (char*)d_ws;
    size_t off = 0;
    unsigned short* W1b  = (unsigned short*)(ws + off); off += (size_t)G_SZ * D_SZ * H_SZ * 2;
    unsigned short* W2b  = (unsigned short*)(ws + off); off += (size_t)G_SZ * H_SZ * D_SZ * 2;
    unsigned short* Waxb = (unsigned short*)(ws + off); off += (size_t)D_SZ * H_SZ * 2;
    unsigned short* abuf = (unsigned short*)(ws + off); off += (size_t)B_SZ * G_SZ * H_SZ * 2;
    float* outbuf = (float*)(ws + off); off += (size_t)B_SZ * G_SZ * D_SZ * 4;
    float* ae     = (float*)(ws + off); off += (size_t)G_SZ * H_SZ * 4;
    float* scores = (float*)(ws + off); off += (size_t)B_SZ * G_SZ * 4;

    {
        size_t n1 = (size_t)G_SZ * D_SZ * H_SZ;
        swizzle_w_kernel<<<(unsigned)((n1 + 255) / 256), 256, 0, stream>>>(W1, W1b, G_SZ, D_SZ, H_SZ);
        swizzle_w_kernel<<<(unsigned)((n1 + 255) / 256), 256, 0, stream>>>(W2, W2b, G_SZ, H_SZ, D_SZ);
        size_t n3 = (size_t)D_SZ * H_SZ;
        swizzle_w_kernel<<<(unsigned)((n3 + 255) / 256), 256, 0, stream>>>(Wa1, Waxb, 1, D_SZ, H_SZ);
    }
    ae_kernel<<<(G_SZ * H_SZ + 255) / 256, 256, 0, stream>>>(emb, Wa1, ba1, ae);

    dim3 g1(B_SZ / 32, G_SZ);
    gemm1_ln_kernel<<<g1, 256, 32 * D_SZ * 2 + 32 * H_SZ * 2, stream>>>(x, W1b, b1, ln_g, ln_b, abuf);

    dim3 g2(B_SZ / 32, 4, G_SZ);
    gemm2_kernel<<<g2, 256, 32 * H_SZ * 2, stream>>>(abuf, W2b, b2, outbuf);

    attn_scores_kernel<<<B_SZ / 32, 256, AT_SMEM, stream>>>(
        x, Waxb, ae, Wa2, ba2, scores);

    combine_kernel<<<B_SZ, 256, 0, stream>>>(scores, outbuf, out);
}
